// SubwordModernBert_52733608460467
// MI455X (gfx1250) — compile-verified, matrix operand order checked by reading, not run on hardware
//
#include <hip/hip_runtime.h>
#include <hip/hip_bf16.h>

typedef __attribute__((ext_vector_type(16))) _Float16 v16h;
typedef __attribute__((ext_vector_type(8)))  _Float16 v8h;
typedef __attribute__((ext_vector_type(8)))  float    v8f;

#define B_   16
#define S_   512
#define H_   768
#define W_   256
#define V_   32000
#define ROWS 4096            // B_*W_
#define KCH  24              // 768/32 K-chunks
#define VT   2000            // 32000/16 column tiles
#define CSPL 8               // column splits (grid.y)
#define TPC  250             // tiles per column split
#define MB   64              // rows per workgroup (4 row-tiles)

// ---------------------------------------------------------------------------
// Kernel 1: segment-mean pool -> (a) f16 A-fragment packed layout for WMMA
//                                (b) plain f32 row-major copy for target dots
// grid: ROWS blocks of 128 threads; block = one (b, w) pooled row
// ---------------------------------------------------------------------------
__global__ void pool_pack_kernel(const float* __restrict__ hs,
                                 const int* __restrict__ word_ids,
                                 _Float16* __restrict__ apk,
                                 float* __restrict__ pooled32) {
    const int row = blockIdx.x;
    const int b = row >> 8, w = row & 255;
    const int* wid = word_ids + b * S_;
    int lo = 0, hi = S_;
    while (lo < hi) { int mid = (lo + hi) >> 1; if (wid[mid] < w) lo = mid + 1; else hi = mid; }
    const int start = lo;
    hi = S_;
    while (lo < hi) { int mid = (lo + hi) >> 1; if (wid[mid] <= w) lo = mid + 1; else hi = mid; }
    const int end = lo;
    const int cnt = end - start;
    const float inv = cnt > 0 ? 1.0f / (float)cnt : 0.0f;

    const int rb = row >> 6, rr = row & 63, rt = rr >> 4, m = rr & 15;
    _Float16* out = apk + (size_t)rb * (MB * H_);

    for (int h = threadIdx.x; h < H_; h += 128) {
        float s = 0.0f;
        for (int si = start; si < end; ++si)
            s += hs[((size_t)b * S_ + si) * H_ + h];
        const float mean = s * inv;                 // cnt==0 -> 0 (matches ref)
        pooled32[(size_t)row * H_ + h] = mean;
        const int c = h >> 5, ko = h & 31;
        const int half = (ko >> 3) & 1;
        const int j = ((ko >> 4) << 3) | (ko & 7);
        const int lane = m + half * 16;
        out[((rt * KCH + c) * 512) + lane * 16 + j] = (_Float16)mean;
    }
}

// ---------------------------------------------------------------------------
// Kernel 2: W_cls fp32 -> f16 packed B-fragment tiles
// grid: (VT, KCH), 64 threads; thread g handles (lane = g>>1, jb = g&1): 8 f16
// ---------------------------------------------------------------------------
__global__ void wpack_kernel(const float* __restrict__ Wc,
                             _Float16* __restrict__ wpk) {
    const int t = blockIdx.x, c = blockIdx.y;
    const int g = threadIdx.x;
    const int lane = g >> 1, jb = g & 1;
    const int nl = lane & 15, half = lane >> 4;
    v8h out8;
#pragma unroll
    for (int i = 0; i < 8; ++i) {
        const int K = jb * 16 + half * 8 + i;      // ISA 16-bit B-fragment K map
        out8[i] = (_Float16)Wc[(size_t)(32 * c + K) * V_ + 16 * t + nl];
    }
    _Float16* dst = wpk + ((size_t)(t * KCH + c) * 512) + lane * 16 + jb * 8;
    *(v8h*)dst = out8;
}

// ---------------------------------------------------------------------------
// Kernel 3: target logit = dot(pooled_row, W_cls[:, tgt]) + b[tgt]  (f32)
// grid: ROWS blocks of 128 threads (3.1 MFLOP total — noise, L2-resident)
// ---------------------------------------------------------------------------
__global__ void tgt_logit_kernel(const float* __restrict__ pooled32,
                                 const float* __restrict__ Wc,
                                 const float* __restrict__ bias,
                                 const int* __restrict__ targets,
                                 float* __restrict__ tl) {
    __shared__ float red[128];
    const int r = blockIdx.x;
    const int tgt = targets[r];
    const float* pr = pooled32 + (size_t)r * H_;
    float s = 0.0f;
    for (int h = threadIdx.x; h < H_; h += 128)
        s += pr[h] * Wc[(size_t)h * V_ + tgt];
    red[threadIdx.x] = s;
    __syncthreads();
    for (int st = 64; st > 0; st >>= 1) {
        if (threadIdx.x < st) red[threadIdx.x] += red[threadIdx.x + st];
        __syncthreads();
    }
    if (threadIdx.x == 0) tl[r] = red[0] + bias[tgt];
}

// ---------------------------------------------------------------------------
// Kernel 4: fused GEMM (WMMA f16->f32) + per-lane online softmax
// grid: (64 rowblocks, CSPL), 256 threads = 8 waves (wave32)
// ---------------------------------------------------------------------------
__global__ void gemm_softmax_kernel(const _Float16* __restrict__ apk,
                                    const _Float16* __restrict__ wpk,
                                    const float* __restrict__ bias,
                                    float* __restrict__ gMax,
                                    float* __restrict__ gSum) {
    __shared__ _Float16 sA[MB * H_];               // 96 KB, packed fragment order
    __shared__ float    sMax[8 * MB];              // per-wave row stats
    __shared__ float    sSum[8 * MB];

    const int tid  = threadIdx.x;
    const int wave = tid >> 5;                     // wave32
    const int lane = tid & 31;
    const int half = lane >> 4, nlane = lane & 15;
    const int rb   = blockIdx.x;
    const int cs   = blockIdx.y;

    // cooperative A-block load (global packed -> LDS, same layout)
    {
        const uint4* src = (const uint4*)(apk + (size_t)rb * (MB * H_));
        uint4* dst = (uint4*)sA;
#pragma unroll 4
        for (int i = tid; i < (MB * H_ * 2) / 16; i += 256) dst[i] = src[i];
    }
    __syncthreads();

    // per-lane online softmax state over this lane's own column stream
    float rmax[32], rsum[32];
#pragma unroll
    for (int i = 0; i < 32; ++i) { rmax[i] = -__builtin_inff(); rsum[i] = 0.0f; }

    const int t0 = cs * TPC;
#pragma unroll 1
    for (int t = t0 + wave; t < t0 + TPC; t += 8) {
        const float bv = bias[t * 16 + nlane];

        v8f acc[4];
#pragma unroll
        for (int rt = 0; rt < 4; ++rt)
#pragma unroll
            for (int r = 0; r < 8; ++r) acc[rt][r] = 0.0f;

        const _Float16* bp = wpk + ((size_t)t * KCH) * 512 + lane * 16;
        // CRITICAL: keep rolled — full unroll spills 96 A-fragments to scratch.
#pragma unroll 1
        for (int c = 0; c < KCH; ++c) {
            const v16h bf = *(const v16h*)(bp + c * 512);
#pragma unroll
            for (int rt = 0; rt < 4; ++rt) {
                const v16h af = *(const v16h*)(sA + (rt * KCH + c) * 512 + lane * 16);
                acc[rt] = __builtin_amdgcn_wmma_f32_16x16x32_f16(
                    false, af, false, bf, (short)0, acc[rt], false, false);
            }
        }

        // epilogue: branchless per-lane online update — 1 exp + ~5 VALU/elem,
        // no shuffles, no stores, no exec-mask traffic
#pragma unroll
        for (int rt = 0; rt < 4; ++rt) {
#pragma unroll
            for (int r = 0; r < 8; ++r) {
                const int i = rt * 8 + r;
                const float v = acc[rt][r] + bv;
                const float d = v - rmax[i];
                const float e = __expf(-fabsf(d));
                const bool gt = d > 0.0f;          // first hit: d = +inf -> gt
                rsum[i] = gt ? rsum[i] * e + 1.0f : rsum[i] + e;
                rmax[i] = gt ? v : rmax[i];
            }
        }
    }

    // ONE cross-lane (16-lane butterfly) merge of (max,sum) per row
#pragma unroll
    for (int i = 0; i < 32; ++i) {
        float m = rmax[i], s = rsum[i];
#pragma unroll
        for (int mask = 1; mask <= 8; mask <<= 1) {
            const float om = __shfl_xor(m, mask, 32);
            const float os = __shfl_xor(s, mask, 32);
            const float nm = fmaxf(m, om);
            s = s * __expf(m - nm) + os * __expf(om - nm);
            m = nm;
        }
        rmax[i] = m; rsum[i] = s;
    }

    // per-wave stats -> LDS (lanes 0 and 16 hold the two row halves)
    if (nlane == 0) {
#pragma unroll
        for (int i = 0; i < 32; ++i) {
            const int rowl = (i >> 3) * 16 + half * 8 + (i & 7);
            sMax[wave * MB + rowl] = rmax[i];
            sSum[wave * MB + rowl] = rsum[i];
        }
    }
    __syncthreads();

    // cross-wave merge -> global per-(row, colsplit) stats
    if (tid < MB) {
        float M = -__builtin_inff();
#pragma unroll
        for (int w = 0; w < 8; ++w) M = fmaxf(M, sMax[w * MB + tid]);
        float Ssum = 0.0f;
#pragma unroll
        for (int w = 0; w < 8; ++w) Ssum += sSum[w * MB + tid] * __expf(sMax[w * MB + tid] - M);
        const int row = rb * MB + tid;
        gMax[row * CSPL + cs] = M;
        gSum[row * CSPL + cs] = Ssum;
    }
}

// ---------------------------------------------------------------------------
// Kernel 5: merge column splits, masked CE mean -> scalar loss
// ---------------------------------------------------------------------------
__global__ void reduce_kernel(const float* __restrict__ gMax,
                              const float* __restrict__ gSum,
                              const float* __restrict__ tl,
                              const int* __restrict__ targets,
                              float* __restrict__ out) {
    __shared__ float ssum[256];
    __shared__ float scnt[256];
    const int tid = threadIdx.x;
    float lsum = 0.0f, lcnt = 0.0f;
    for (int r = tid; r < ROWS; r += 256) {
        float M = -__builtin_inff();
#pragma unroll
        for (int j = 0; j < CSPL; ++j) M = fmaxf(M, gMax[r * CSPL + j]);
        float Ssum = 0.0f;
#pragma unroll
        for (int j = 0; j < CSPL; ++j) Ssum += gSum[r * CSPL + j] * __expf(gMax[r * CSPL + j] - M);
        const float lse = M + __logf(Ssum);
        const int t = targets[r];
        if (t != 0) { lsum += (lse - tl[r]); lcnt += 1.0f; }
    }
    ssum[tid] = lsum; scnt[tid] = lcnt;
    __syncthreads();
    for (int s = 128; s > 0; s >>= 1) {
        if (tid < s) { ssum[tid] += ssum[tid + s]; scnt[tid] += scnt[tid + s]; }
        __syncthreads();
    }
    if (tid == 0) out[0] = ssum[0] / fmaxf(scnt[0], 1.0f);
}

// ---------------------------------------------------------------------------
extern "C" void kernel_launch(void* const* d_in, const int* in_sizes, int n_in,
                              void* d_out, int out_size, void* d_ws, size_t ws_size,
                              hipStream_t stream) {
    const float* hs  = (const float*)d_in[0];   // (16,512,768) f32
    const int*   wid = (const int*)d_in[1];     // (16,512) i32
    const int*   tgt = (const int*)d_in[2];     // (16,256) i32
    const float* Wc  = (const float*)d_in[3];   // (768,32000) f32
    const float* bc  = (const float*)d_in[4];   // (32000,) f32
    float* out = (float*)d_out;
    char* ws = (char*)d_ws;

    _Float16* wpk  = (_Float16*)ws;                      // 49,152,000 B
    _Float16* apk  = (_Float16*)(ws + 49152000);         //  6,291,456 B
    float*    p32  = (float*)(ws + 55443456);            // 12,582,912 B
    float*    gMax = (float*)(ws + 68026368);            //    131,072 B
    float*    gSum = (float*)(ws + 68157440);            //    131,072 B
    float*    tl   = (float*)(ws + 68288512);            //     16,384 B

    pool_pack_kernel<<<ROWS, 128, 0, stream>>>(hs, wid, apk, p32);
    wpack_kernel<<<dim3(VT, KCH), 64, 0, stream>>>(Wc, wpk);
    tgt_logit_kernel<<<ROWS, 128, 0, stream>>>(p32, Wc, bc, tgt, tl);
    gemm_softmax_kernel<<<dim3(ROWS / MB, CSPL), 256, 0, stream>>>(
        apk, wpk, bc, gMax, gSum);
    reduce_kernel<<<1, 256, 0, stream>>>(gMax, gSum, tl, tgt, out);
}
